// GNN25_27410481283394
// MI455X (gfx1250) — matrix operand order repaired and checked by reading, not compiled, not run on hardware
//
#include <hip/hip_runtime.h>
#include <hip/hip_bf16.h>
#include <math.h>

typedef __attribute__((ext_vector_type(16))) _Float16 v16h;
typedef __attribute__((ext_vector_type(8)))  _Float16 v8h;
typedef __attribute__((ext_vector_type(8)))  float    v8f;

#define NEG_SLOPE 0.2f
#define NHEADS 6

// ---------------------------------------------------------------------------
// ordered-uint encoding so unsigned atomicMax == float max (handles negatives)
// ---------------------------------------------------------------------------
__device__ __forceinline__ unsigned f2ord(float f) {
    unsigned u = __float_as_uint(f);
    return (u & 0x80000000u) ? ~u : (u | 0x80000000u);
}
__device__ __forceinline__ float ord2f(unsigned u) {
    return (u & 0x80000000u) ? __uint_as_float(u & 0x7fffffffu)
                             : __uint_as_float(~u);
}

// ---------------------------------------------------------------------------
// Precision staging: convert f32 activations -> f16, zero-padded K -> Kp.
// A_f16 is [Mp x Kp] row-major.  Coalesced along Kp.
// ---------------------------------------------------------------------------
__global__ void gat_cvt_a(const float* __restrict__ A, _Float16* __restrict__ Ah,
                          int M, int K, int Mp, int Kp) {
    size_t i = (size_t)blockIdx.x * blockDim.x + threadIdx.x;
    if (i >= (size_t)Mp * Kp) return;
    const int m = (int)(i / Kp);
    const int k = (int)(i - (size_t)m * Kp);
    float v = 0.0f;
    if (m < M && k < K) v = A[(size_t)m * K + k];
    Ah[i] = (_Float16)v;
}

// Weights: W is [K x C] row-major (from [F_in, H, F]); produce column-major
// f16 Bt [C x Kp] so every WMMA B-fragment is one contiguous 32B load.
__global__ void gat_cvt_bt(const float* __restrict__ W, _Float16* __restrict__ Bt,
                           int K, int C, int Kp) {
    size_t i = (size_t)blockIdx.x * blockDim.x + threadIdx.x;
    if (i >= (size_t)C * Kp) return;
    const int n = (int)(i / Kp);
    const int k = (int)(i - (size_t)n * Kp);
    float v = 0.0f;
    if (k < K) v = W[(size_t)k * C + n];
    Bt[i] = (_Float16)v;
}

// ---------------------------------------------------------------------------
// WMMA GEMM: C[M,Nc] = Ah[Mp,Kp] @ B, with B given transposed as Bt[Nc,Kp].
// One wave computes a 16x32 strip of C: one shared A fragment, two B
// fragments, two v_wmma_f32_16x16x32_f16 per 32-wide K step.  All fragment
// loads are contiguous wide loads (no branches in the hot loop).
// Fragment layouts (wave32); A/B use this code's contiguous k order, not the ISA table's:
//   A 16x32 f16 : lane l -> row m=l%16, half h=l/16;
//                 elem i = K(16h+i)  (contiguous in Kp)
//   B 32x16 f16 : lane l -> col n=l%16 ; elem i = K(16h+i)  (contiguous in Kp)
//   C/D 16x16 f32: lane l -> col n=l%16 ; vgpr v -> row m = v + 8h
// ---------------------------------------------------------------------------
__global__ void gat_wmma_gemm(const _Float16* __restrict__ Ah,
                              const _Float16* __restrict__ Bt,
                              float* __restrict__ C,
                              int M, int Mp, int Kp, int Nc) {
    const int lane = threadIdx.x & 31;
    const int wave = threadIdx.x >> 5;
    const int groupsN = Nc >> 5;                 // 32 output cols per wave
    const int tilesM  = Mp >> 4;
    const int wid = blockIdx.x * (blockDim.x >> 5) + wave;
    if (wid >= tilesM * groupsN) return;

    const int tm = wid / groupsN;
    const int gn = wid - tm * groupsN;
    const int row0 = tm << 4;
    const int col0 = gn << 5;
    const int half = lane >> 4;                  // 0 or 1
    const int l16  = lane & 15;
    const int mr   = row0 + l16;

    const _Float16* arow = Ah + (size_t)mr * Kp;
    const _Float16* bcol0 = Bt + (size_t)(col0 + l16) * Kp;        // group 0
    const _Float16* bcol1 = Bt + (size_t)(col0 + 16 + l16) * Kp;   // group 1

    v8f acc0 = {};
    v8f acc1 = {};
    for (int k0 = 0; k0 < Kp; k0 += 32) {
        // A fragment: two contiguous 16B chunks
        const v8h a_lo = *(const v8h*)(arow + k0 + 16 * half);
        const v8h a_hi = *(const v8h*)(arow + k0 + 16 * half + 8);
        v16h a;
        #pragma unroll
        for (int i = 0; i < 8; ++i) { a[i] = a_lo[i]; a[8 + i] = a_hi[i]; }
        // B fragments: one contiguous 32B chunk each
        const v16h b0 = *(const v16h*)(bcol0 + k0 + 16 * half);
        const v16h b1 = *(const v16h*)(bcol1 + k0 + 16 * half);
        acc0 = __builtin_amdgcn_wmma_f32_16x16x32_f16(
                   false, a, false, b0, (short)0, acc0, false, false);
        acc1 = __builtin_amdgcn_wmma_f32_16x16x32_f16(
                   false, a, false, b1, (short)0, acc1, false, false);
    }
    // store D (16 rows x 32 cols)
    #pragma unroll
    for (int v = 0; v < 8; ++v) {
        const int m = row0 + v + 8 * half;
        if (m < M) {
            float* crow = C + (size_t)m * Nc + col0;
            crow[l16]      = acc0[v];
            crow[16 + l16] = acc1[v];
        }
    }
}

// ---------------------------------------------------------------------------
// Per-node attention logits: es[n,h] = <h[n,h,:], a_src[h,:]>, ed likewise.
// h laid out [N, H*F] row-major (head-major inner).
// ---------------------------------------------------------------------------
__global__ void gat_node_scores(const float* __restrict__ h,
                                const float* __restrict__ a_src,
                                const float* __restrict__ a_dst,
                                float* __restrict__ es,
                                float* __restrict__ ed,
                                int N, int F) {
    const int i = blockIdx.x * blockDim.x + threadIdx.x;
    if (i >= N * NHEADS) return;
    const int n  = i / NHEADS;
    const int hd = i - n * NHEADS;
    const float* hp = h + (size_t)n * NHEADS * F + (size_t)hd * F;
    const float* as = a_src + (size_t)hd * F;
    const float* ad = a_dst + (size_t)hd * F;
    float s = 0.0f, d = 0.0f;
    for (int f = 0; f < F; ++f) {
        const float v = hp[f];
        s = fmaf(v, as[f], s);
        d = fmaf(v, ad[f], d);
    }
    es[i] = s;
    ed[i] = d;
}

// ---------------------------------------------------------------------------
// init helpers
// ---------------------------------------------------------------------------
__global__ void gat_fill_u32(unsigned* __restrict__ p, unsigned v, size_t n) {
    size_t i = (size_t)blockIdx.x * blockDim.x + threadIdx.x;
    if (i < n) p[i] = v;
}
__global__ void gat_fill_f32(float* __restrict__ p, float v, size_t n) {
    size_t i = (size_t)blockIdx.x * blockDim.x + threadIdx.x;
    if (i < n) p[i] = v;
}

// ---------------------------------------------------------------------------
// Pass 1 over edges: segment max of leaky_relu(es[src]+ed[dst]) into mx[dst,h]
// ---------------------------------------------------------------------------
__global__ void gat_edge_max(const long long* __restrict__ src,
                             const long long* __restrict__ dst,
                             const float* __restrict__ es,
                             const float* __restrict__ ed,
                             unsigned* __restrict__ mx,
                             long long EH) {
    const long long i = (long long)blockIdx.x * blockDim.x + threadIdx.x;
    if (i >= EH) return;
    const long long e = i / NHEADS;
    const int hd = (int)(i - e * NHEADS);
    __builtin_prefetch(&src[e + 2048], 0, 1);   // global_prefetch_b8
    const int s = (int)src[e];
    const int d = (int)dst[e];
    float v = es[s * NHEADS + hd] + ed[d * NHEADS + hd];
    v = (v > 0.0f) ? v : NEG_SLOPE * v;
    atomicMax(&mx[d * NHEADS + hd], f2ord(v));
}

// ---------------------------------------------------------------------------
// Pass 2 over edges: denom[dst,h] += exp(e - max[dst,h])
// ---------------------------------------------------------------------------
__global__ void gat_edge_expsum(const long long* __restrict__ src,
                                const long long* __restrict__ dst,
                                const float* __restrict__ es,
                                const float* __restrict__ ed,
                                const unsigned* __restrict__ mx,
                                float* __restrict__ den,
                                long long EH) {
    const long long i = (long long)blockIdx.x * blockDim.x + threadIdx.x;
    if (i >= EH) return;
    const long long e = i / NHEADS;
    const int hd = (int)(i - e * NHEADS);
    __builtin_prefetch(&dst[e + 2048], 0, 1);
    const int s = (int)src[e];
    const int d = (int)dst[e];
    float v = es[s * NHEADS + hd] + ed[d * NHEADS + hd];
    v = (v > 0.0f) ? v : NEG_SLOPE * v;
    const float w = __expf(v - ord2f(mx[d * NHEADS + hd]));
    atomicAdd(&den[d * NHEADS + hd], w);
}

// ---------------------------------------------------------------------------
// Pass 3 over (edge, head): out[dst, h*F + f] += att * hfeat[src, h*F + f]
// hfeat rows (<=153.6 MB total) are L2-resident on MI455X (192 MB L2),
// so the random src gathers and dst atomics both resolve in L2.
// ---------------------------------------------------------------------------
__global__ void gat_edge_aggregate(const long long* __restrict__ src,
                                   const long long* __restrict__ dst,
                                   const float* __restrict__ es,
                                   const float* __restrict__ ed,
                                   const unsigned* __restrict__ mx,
                                   const float* __restrict__ den,
                                   const float* __restrict__ h,
                                   float* __restrict__ out,
                                   long long EH, int F) {
    const long long i = (long long)blockIdx.x * blockDim.x + threadIdx.x;
    if (i >= EH) return;
    const long long e = i / NHEADS;
    const int hd = (int)(i - e * NHEADS);
    const int s = (int)src[e];
    const int d = (int)dst[e];
    float v = es[s * NHEADS + hd] + ed[d * NHEADS + hd];
    v = (v > 0.0f) ? v : NEG_SLOPE * v;
    const int C = NHEADS * F;
    const float w   = __expf(v - ord2f(mx[d * NHEADS + hd]));
    const float att = w / (den[d * NHEADS + hd] + 1e-16f);
    const float* hp = h   + (size_t)s * C + (size_t)hd * F;
    float*       op = out + (size_t)d * C + (size_t)hd * F;
    __builtin_prefetch(hp, 0, 1);
    #pragma unroll 4
    for (int f = 0; f < F; ++f) {
        atomicAdd(&op[f], att * hp[f]);
    }
}

// ---------------------------------------------------------------------------
// ELU in place over [N*C]
// ---------------------------------------------------------------------------
__global__ void gat_elu(float* __restrict__ p, size_t n) {
    size_t i = (size_t)blockIdx.x * blockDim.x + threadIdx.x;
    if (i >= n) return;
    const float v = p[i];
    p[i] = (v > 0.0f) ? v : (__expf(v) - 1.0f);
}

// ---------------------------------------------------------------------------
// Sum pool: one block per column (384 blocks), deterministic, no atomics.
// ---------------------------------------------------------------------------
__global__ void gat_pool(const float* __restrict__ h, float* __restrict__ pooled,
                         int N, int C) {
    __shared__ float red[256];
    const int c = blockIdx.x;
    float s = 0.0f;
    for (int n = threadIdx.x; n < N; n += blockDim.x)
        s += h[(size_t)n * C + c];
    red[threadIdx.x] = s;
    __syncthreads();
    for (int off = 128; off > 0; off >>= 1) {
        if (threadIdx.x < off) red[threadIdx.x] += red[threadIdx.x + off];
        __syncthreads();
    }
    if (threadIdx.x == 0) pooled[c] = red[0];
}

// ---------------------------------------------------------------------------
// Finalize: L2-normalize pooled[384], dot with Wd, add bd -> out[0]
// ---------------------------------------------------------------------------
__global__ void gat_finalize(const float* __restrict__ pooled,
                             const float* __restrict__ Wd,
                             const float* __restrict__ bd,
                             float* __restrict__ out) {
    __shared__ float red[384];
    __shared__ float inv_norm;
    const int t = threadIdx.x;              // 0..383
    const float v = pooled[t];
    red[t] = v * v;
    __syncthreads();
    if (t == 0) {
        float ss = 0.0f;
        for (int i = 0; i < 384; ++i) ss += red[i];
        inv_norm = 1.0f / fmaxf(sqrtf(ss), 1e-12f);
    }
    __syncthreads();
    red[t] = v * inv_norm * Wd[t];
    __syncthreads();
    if (t == 0) {
        float s = 0.0f;
        for (int i = 0; i < 384; ++i) s += red[i];
        out[0] = s + bd[0];
    }
}

// ---------------------------------------------------------------------------
// Host-side layer driver
// ---------------------------------------------------------------------------
static void run_gat_layer(const float* x, int K,
                          const float* W, const float* a_s, const float* a_d,
                          int F,
                          const long long* src, const long long* dst,
                          long long E, int N,
                          _Float16* aF16, _Float16* bF16,
                          float* hbuf, float* aggbuf,
                          float* es, float* ed, unsigned* mx, float* den,
                          hipStream_t stream) {
    const int C  = NHEADS * F;
    const int Kp = (K + 31) & ~31;
    const int Mp = (N + 15) & ~15;

    // 0) stage A (f32->f16, pad) and Bt (transpose+f16, pad)
    {
        const size_t na = (size_t)Mp * Kp;
        gat_cvt_a<<<(unsigned)((na + 255) / 256), 256, 0, stream>>>(
            x, aF16, N, K, Mp, Kp);
        const size_t nb = (size_t)C * Kp;
        gat_cvt_bt<<<(unsigned)((nb + 255) / 256), 256, 0, stream>>>(
            W, bF16, K, C, Kp);
    }
    // 1) h = x @ W via WMMA (16x32 strip per wave)
    {
        const int waves  = (Mp / 16) * (C / 32);
        const int blocks = (waves + 7) / 8;       // 8 waves per 256-thr block
        gat_wmma_gemm<<<blocks, 256, 0, stream>>>(aF16, bF16, hbuf, N, Mp, Kp, C);
    }
    // 2) per-node logits
    {
        const int nh = N * NHEADS;
        gat_node_scores<<<(nh + 255) / 256, 256, 0, stream>>>(hbuf, a_s, a_d,
                                                              es, ed, N, F);
    }
    // 3) init mx (-inf ordered), den (0), agg (0)
    {
        const size_t nh = (size_t)N * NHEADS;
        gat_fill_u32<<<(unsigned)((nh + 255) / 256), 256, 0, stream>>>(
            mx, 0x007FFFFFu /* ordered(-inf) */, nh);
        gat_fill_f32<<<(unsigned)((nh + 255) / 256), 256, 0, stream>>>(den, 0.0f, nh);
        const size_t nc = (size_t)N * C;
        gat_fill_f32<<<(unsigned)((nc + 255) / 256), 256, 0, stream>>>(aggbuf, 0.0f, nc);
    }
    // 4) softmax max / sum / aggregate over edges
    {
        const long long EH = E * NHEADS;
        const unsigned blocks = (unsigned)((EH + 255) / 256);
        gat_edge_max<<<blocks, 256, 0, stream>>>(src, dst, es, ed, mx, EH);
        gat_edge_expsum<<<blocks, 256, 0, stream>>>(src, dst, es, ed, mx, den, EH);
        gat_edge_aggregate<<<blocks, 256, 0, stream>>>(src, dst, es, ed, mx, den,
                                                       hbuf, aggbuf, EH, F);
    }
    // 5) ELU in place -> becomes next layer input
    {
        const size_t nc = (size_t)N * C;
        gat_elu<<<(unsigned)((nc + 255) / 256), 256, 0, stream>>>(aggbuf, nc);
    }
}

extern "C" void kernel_launch(void* const* d_in, const int* in_sizes, int n_in,
                              void* d_out, int out_size, void* d_ws, size_t ws_size,
                              hipStream_t stream) {
    const float*     x   = (const float*)d_in[0];
    const long long* ei  = (const long long*)d_in[1];
    const float*     W1  = (const float*)d_in[2];
    const float*     a1s = (const float*)d_in[3];
    const float*     a1d = (const float*)d_in[4];
    const float*     W2  = (const float*)d_in[5];
    const float*     a2s = (const float*)d_in[6];
    const float*     a2d = (const float*)d_in[7];
    const float*     W3  = (const float*)d_in[8];
    const float*     a3s = (const float*)d_in[9];
    const float*     a3d = (const float*)d_in[10];
    const float*     Wd  = (const float*)d_in[11];
    const float*     bd  = (const float*)d_in[12];
    float* out = (float*)d_out;

    const int       N = in_sizes[0] / 11;
    const long long E = (long long)in_sizes[1] / 2;
    const long long* src = ei;
    const long long* dst = ei + E;
    const int Mp = (N + 15) & ~15;

    // ---- workspace partition (256B aligned) ----
    size_t off = 0;
    auto take = [&](size_t bytes) -> char* {
        char* p = (char*)d_ws + off;
        off += (bytes + 255) & ~(size_t)255;
        return p;
    };
    float*     bufA   = (float*)take((size_t)N * 384 * sizeof(float)); // h feats
    float*     bufB   = (float*)take((size_t)N * 384 * sizeof(float)); // agg/input
    _Float16*  aF16   = (_Float16*)take((size_t)Mp * 192 * sizeof(_Float16));
    _Float16*  bF16   = (_Float16*)take((size_t)384 * 192 * sizeof(_Float16));
    float*     es     = (float*)take((size_t)N * NHEADS * sizeof(float));
    float*     edv    = (float*)take((size_t)N * NHEADS * sizeof(float));
    unsigned*  mx     = (unsigned*)take((size_t)N * NHEADS * sizeof(unsigned));
    float*     den    = (float*)take((size_t)N * NHEADS * sizeof(float));
    float*     pooled = (float*)take(384 * sizeof(float));
    (void)ws_size; (void)n_in; (void)out_size;

    // Layer 1: [N,11] -> h [N,96] -> agg+ELU in bufB
    run_gat_layer(x,    11, W1, a1s, a1d, 16, src, dst, E, N,
                  aF16, bF16, bufA, bufB, es, edv, mx, den, stream);
    // Layer 2: bufB [N,96] -> h [N,192] (bufA) -> agg+ELU back into bufB
    run_gat_layer(bufB, 96, W2, a2s, a2d, 32, src, dst, E, N,
                  aF16, bF16, bufA, bufB, es, edv, mx, den, stream);
    // Layer 3: bufB [N,192] -> h [N,384] (bufA) -> agg+ELU back into bufB
    run_gat_layer(bufB, 192, W3, a3s, a3d, 64, src, dst, E, N,
                  aF16, bF16, bufA, bufB, es, edv, mx, den, stream);

    // Sum pool (deterministic, one block per column), normalize + dense
    gat_pool<<<384, 256, 0, stream>>>(bufB, pooled, N, 384);
    gat_finalize<<<1, 384, 0, stream>>>(pooled, Wd, bd, out);
}
